// RSlot_module_34144990003584
// MI455X (gfx1250) — matrix operand order repaired and checked by reading, not compiled, not run on hardware
//
#include <hip/hip_runtime.h>
#include <hip/hip_bf16.h>

typedef __attribute__((ext_vector_type(16))) __bf16 v16bf;
typedef __attribute__((ext_vector_type(8)))  __bf16 v8bf;
typedef __attribute__((ext_vector_type(4)))  __bf16 v4bf;
typedef __attribute__((ext_vector_type(8)))  float  v8f;

#define BM 64
#define BN 64
#define BK 32

// ---------------------------------------------------------------------------
// Generic bf16-WMMA GEMM with f32 accumulate.
//   C[b][m][n] (+)= sum_tap sum_k A(b,tap,m,k) * B(b,tap,k,n)  (+ bias[m]) (relu)
// A element: A + b*sAb + tap*sAtap + m*sAm + k*sAk   (f32, converted to bf16)
// B element: B + b*sBb + tap*sBtap + k*sBk + n*sBt
// flags: bit0 = accumulate onto existing C, bit1 = relu
// Requires: M % 64 == 0, N % 64 == 0, K % 32 == 0. 256 threads (8 waves).
// ---------------------------------------------------------------------------
__global__ __launch_bounds__(256) void gemm_bf16_wmma(
    const float* __restrict__ A, long sAm, long sAk, long sAtap, long sAb,
    const float* __restrict__ B, long sBk, long sBt, long sBtap, long sBb,
    float* __restrict__ C, long ldc, long sCb,
    const float* __restrict__ bias,
    int K, int ntaps, int flags)
{
    __shared__ __align__(16) __bf16 As[BM * BK];      // [m][k]
    __shared__ __align__(16) __bf16 Bs[BN * BK];      // [n][k] (transposed)

    const int tid  = threadIdx.x;
    const int m0   = blockIdx.y * BM;
    const int n0   = blockIdx.x * BN;
    const int b    = blockIdx.z;
    const int wave = tid >> 5, lane = tid & 31;
    const int wr   = wave >> 1;        // 0..3 : 16-row strip
    const int wc   = wave & 1;         // 0..1 : 32-col strip (two 16x16 tiles)
    const int half = lane >> 4, lm = lane & 15;

    v8f acc0 = {}; v8f acc1 = {};
    const float* Ab = A + (long)b * sAb;
    const float* Bb = B + (long)b * sBb;

    for (int tap = 0; tap < ntaps; ++tap) {
        const float* At = Ab + (long)tap * sAtap;
        const float* Bt = Bb + (long)tap * sBtap;
        // Uniform-per-block path selection (no divergence around WMMA).
        const bool aVecK = (sAk == 1) && ((sAm & 3) == 0) &&
                           ((((unsigned long)At) & 15) == 0);
        const bool aVecM = (sAm == 1) && ((sAk & 3) == 0) &&
                           ((((unsigned long)At) & 15) == 0);
        const bool bVecT = (sBt == 1) && ((sBk & 3) == 0) &&
                           ((((unsigned long)Bt) & 15) == 0);

        for (int kb = 0; kb < K; kb += BK) {
            // ---------------- A tile: 64 x 32 ----------------
            if (aVecK) {
#pragma unroll
                for (int i = 0; i < 2; ++i) {
                    int vidx = tid + i * 256;          // 512 vec4 loads
                    int r = vidx >> 3, kq = (vidx & 7) << 2;
                    float4 f = *(const float4*)(At + (long)(m0 + r) * sAm + (kb + kq));
                    v4bf h;
                    h[0] = (__bf16)f.x; h[1] = (__bf16)f.y;
                    h[2] = (__bf16)f.z; h[3] = (__bf16)f.w;
                    *(v4bf*)&As[r * BK + kq] = h;      // 8B ds store
                }
            } else if (aVecM) {
#pragma unroll
                for (int i = 0; i < 2; ++i) {
                    int vidx = tid + i * 256;
                    int kk = vidx >> 4, mq = (vidx & 15) << 2;
                    float4 f = *(const float4*)(At + (m0 + mq) + (long)(kb + kk) * sAk);
                    As[(mq + 0) * BK + kk] = (__bf16)f.x;
                    As[(mq + 1) * BK + kk] = (__bf16)f.y;
                    As[(mq + 2) * BK + kk] = (__bf16)f.z;
                    As[(mq + 3) * BK + kk] = (__bf16)f.w;
                }
            } else {
                float tmp[8];
#pragma unroll
                for (int i = 0; i < 8; ++i) {
                    int idx = tid + i * 256;
                    int r = idx >> 5, kk = idx & 31;
                    tmp[i] = At[(long)(m0 + r) * sAm + (long)(kb + kk) * sAk];
                }
#pragma unroll
                for (int i = 0; i < 8; ++i) As[tid + i * 256] = (__bf16)tmp[i];
            }
            // ---------------- B tile: 32 x 64 (stored transposed) -----------
            if (bVecT) {
#pragma unroll
                for (int i = 0; i < 2; ++i) {
                    int vidx = tid + i * 256;
                    int kk = vidx >> 4, tq = (vidx & 15) << 2;
                    float4 f = *(const float4*)(Bt + (long)(kb + kk) * sBk + (n0 + tq));
                    Bs[(tq + 0) * BK + kk] = (__bf16)f.x;
                    Bs[(tq + 1) * BK + kk] = (__bf16)f.y;
                    Bs[(tq + 2) * BK + kk] = (__bf16)f.z;
                    Bs[(tq + 3) * BK + kk] = (__bf16)f.w;
                }
            } else {
                float tmp[8];
#pragma unroll
                for (int i = 0; i < 8; ++i) {
                    int idx = tid + i * 256;
                    int kk = idx >> 6, tt = idx & 63;
                    tmp[i] = Bt[(long)(kb + kk) * sBk + (long)(n0 + tt) * sBt];
                }
#pragma unroll
                for (int i = 0; i < 8; ++i) {
                    int idx = tid + i * 256;
                    int kk = idx >> 6, tt = idx & 63;
                    Bs[tt * BK + kk] = (__bf16)tmp[i];
                }
            }
            if (kb + BK < K) {  // -> global_prefetch_b8 for the next A tile
                __builtin_prefetch(At + (long)m0 * sAm + (long)(kb + BK) * sAk, 0, 3);
            }
            __syncthreads();

            // A fragment (16x32 bf16): lane = M row; lanes 0-15 hold K 0-15;
            // lanes 16-31 hold K 16-31 (same k order as B).
            const int am = wr * 16 + lm;
            v8bf A0 = *(const v8bf*)&As[am * BK + half * 16];
            v8bf A1 = *(const v8bf*)&As[am * BK + half * 16 + 8];
            v16bf af;
#pragma unroll
            for (int e = 0; e < 8; ++e) { af[e] = A0[e]; af[8 + e] = A1[e]; }

#pragma unroll
            for (int j = 0; j < 2; ++j) {
                // B fragment (32x16 bf16): lane%16 = N col; lanes 0-15 K 0-15,
                // lanes 16-31 K 16-31.
                const int bn = wc * 32 + j * 16 + lm;
                v8bf B0 = *(const v8bf*)&Bs[bn * BK + half * 16];
                v8bf B1 = *(const v8bf*)&Bs[bn * BK + half * 16 + 8];
                v16bf bfr;
#pragma unroll
                for (int e = 0; e < 8; ++e) { bfr[e] = B0[e]; bfr[8 + e] = B1[e]; }
                if (j == 0)
                    acc0 = __builtin_amdgcn_wmma_f32_16x16x32_bf16(
                        false, af, false, bfr, (short)0, acc0, false, false);
                else
                    acc1 = __builtin_amdgcn_wmma_f32_16x16x32_bf16(
                        false, af, false, bfr, (short)0, acc1, false, false);
            }
            __syncthreads();
        }
    }

    // Epilogue. C layout (ISA 7.12.2): VGPR i -> row half*8+i, col = lane%16.
    float* Cb = C + (long)b * sCb;
#pragma unroll
    for (int j = 0; j < 2; ++j) {
        const int col = n0 + wc * 32 + j * 16 + lm;
#pragma unroll
        for (int i = 0; i < 8; ++i) {
            const int row = m0 + wr * 16 + half * 8 + i;
            float v = (j == 0) ? acc0[i] : acc1[i];
            long off = (long)row * ldc + col;
            if (flags & 1) v += Cb[off];
            if (bias) v += bias[row];
            if (flags & 2) v = fmaxf(v, 0.f);
            Cb[off] = v;
        }
    }
}

// ---------------------------------------------------------------------------
// Grouped conv1d k=3, groups=32 on [4][128][512], with ReLU (get_convs middle).
// ---------------------------------------------------------------------------
__global__ __launch_bounds__(256) void grouped_conv3_relu(
    const float* __restrict__ in, const float* __restrict__ w,
    const float* __restrict__ bias, float* __restrict__ out)
{
    int idx = blockIdx.x * blockDim.x + threadIdx.x;
    int n = blockIdx.y;
    if (idx >= 128 * 512) return;
    int c = idx >> 9, t = idx & 511;
    int g = c >> 2;
    const float* ip = in + ((long)n * 128 + g * 4) * 512;
    const float* wp = w + c * 12;   // [4][3]
    float s = bias[c];
#pragma unroll
    for (int j = 0; j < 4; ++j) {
        float x0 = (t > 0)   ? ip[j * 512 + t - 1] : 0.f;
        float x1 =             ip[j * 512 + t];
        float x2 = (t < 511) ? ip[j * 512 + t + 1] : 0.f;
        s += wp[j * 3 + 0] * x0 + wp[j * 3 + 1] * x1 + wp[j * 3 + 2] * x2;
    }
    out[((long)n * 128 + c) * 512 + t] = fmaxf(s, 0.f);
}

// ---------------------------------------------------------------------------
// In-place softmax over axis -2 (rows t) per column s of [4][512][512].
// grid = (512 cols, 4 batches), 128 threads.
// ---------------------------------------------------------------------------
__global__ __launch_bounds__(128) void col_softmax(float* __restrict__ M)
{
    int s = blockIdx.x, n = blockIdx.y, tid = threadIdx.x;
    float* col = M + (long)n * 512 * 512 + s;
    __shared__ float red[128];
    float v[4]; float mx = -3.4e38f;
#pragma unroll
    for (int i = 0; i < 4; ++i) { v[i] = col[(long)(tid + i * 128) * 512]; mx = fmaxf(mx, v[i]); }
    red[tid] = mx; __syncthreads();
    for (int st = 64; st > 0; st >>= 1) { if (tid < st) red[tid] = fmaxf(red[tid], red[tid + st]); __syncthreads(); }
    mx = red[0]; __syncthreads();
    float sm = 0.f;
#pragma unroll
    for (int i = 0; i < 4; ++i) { v[i] = __expf(v[i] - mx); sm += v[i]; }
    red[tid] = sm; __syncthreads();
    for (int st = 64; st > 0; st >>= 1) { if (tid < st) red[tid] += red[tid + st]; __syncthreads(); }
    float inv = 1.f / red[0];
#pragma unroll
    for (int i = 0; i < 4; ++i) col[(long)(tid + i * 128) * 512] = v[i] * inv;
}

// ---------------------------------------------------------------------------
// U[n][tap][t] = sum_c agg_w[c][tap] * Y[n][c][t]     (agg_w flat: c*Kw+tap)
// ---------------------------------------------------------------------------
__global__ __launch_bounds__(256) void agg_u(
    const float* __restrict__ Y, const float* __restrict__ aggw,
    float* __restrict__ U, int Kw)
{
    int n = blockIdx.y;
    int idx = blockIdx.x * blockDim.x + threadIdx.x;
    if (idx >= Kw * 512) return;
    int tap = idx >> 9, t = idx & 511;
    const float* Yn = Y + (long)n * 128 * 512;
    float s = 0.f;
    for (int c = 0; c < 128; ++c) s += aggw[c * Kw + tap] * Yn[c * 512 + t];
    U[(long)n * Kw * 512 + idx] = s;
}

// ---------------------------------------------------------------------------
// Banded pre-softmax z + dense column softmax, (+)= into conns[n][t][s].
// z[t,s] = sum_{t' in [t-a,t+a] ∩ [s-a,s+a] ∩ [0,T)} U[t'-t+a][t'] ; 0 elsewhere.
// grid = (512 cols, 4 batches), 128 threads.
// ---------------------------------------------------------------------------
__global__ __launch_bounds__(128) void band_softmax(
    const float* __restrict__ U, float* __restrict__ conns, int a, int accum)
{
    int s = blockIdx.x, n = blockIdx.y, tid = threadIdx.x;
    int Kw = 2 * a + 1;
    const float* Un = U + (long)n * Kw * 512;
    __shared__ float z[512];
    __shared__ float red[128];
    for (int t = tid; t < 512; t += 128) {
        float zz = 0.f;
        int lo = t - a; if (s - a > lo) lo = s - a; if (lo < 0) lo = 0;
        int hi = t + a; if (s + a < hi) hi = s + a; if (hi > 511) hi = 511;
        for (int tp = lo; tp <= hi; ++tp) zz += Un[(tp - t + a) * 512 + tp];
        z[t] = zz;
    }
    __syncthreads();
    float mx = -3.4e38f;
    for (int t = tid; t < 512; t += 128) mx = fmaxf(mx, z[t]);
    red[tid] = mx; __syncthreads();
    for (int st = 64; st > 0; st >>= 1) { if (tid < st) red[tid] = fmaxf(red[tid], red[tid + st]); __syncthreads(); }
    mx = red[0]; __syncthreads();
    float sm = 0.f;
    for (int t = tid; t < 512; t += 128) sm += __expf(z[t] - mx);
    red[tid] = sm; __syncthreads();
    for (int st = 64; st > 0; st >>= 1) { if (tid < st) red[tid] += red[tid + st]; __syncthreads(); }
    float inv = 1.f / red[0];
    float* cn = conns + (long)n * 512 * 512 + s;
    for (int t = tid; t < 512; t += 128) {
        float v = __expf(z[t] - mx) * inv;
        if (accum) v += cn[(long)t * 512];
        cn[(long)t * 512] = v;
    }
}

// ---------------------------------------------------------------------------
// BatchNorm1d (training stats over (N,T), biased var) + ReLU. Block per channel.
// ---------------------------------------------------------------------------
__global__ __launch_bounds__(256) void bn_relu(
    const float* __restrict__ acc, const float* __restrict__ g,
    const float* __restrict__ bb, float* __restrict__ out)
{
    int o = blockIdx.x, tid = threadIdx.x;
    __shared__ float rs[256], rq[256];
    float s = 0.f, q = 0.f;
    for (int n = 0; n < 4; ++n) {
        const float* p = acc + ((long)n * 512 + o) * 512;
        for (int t = tid; t < 512; t += 256) { float v = p[t]; s += v; q += v * v; }
    }
    rs[tid] = s; rq[tid] = q; __syncthreads();
    for (int st = 128; st > 0; st >>= 1) {
        if (tid < st) { rs[tid] += rs[tid + st]; rq[tid] += rq[tid + st]; }
        __syncthreads();
    }
    float mean = rs[0] * (1.f / 2048.f);
    float var  = rq[0] * (1.f / 2048.f) - mean * mean;
    float sc = g[o] * rsqrtf(var + 1e-5f);
    float sh = bb[o] - mean * sc;
    for (int n = 0; n < 4; ++n) {
        const float* p = acc + ((long)n * 512 + o) * 512;
        float* po = out + ((long)n * 512 + o) * 512;
        for (int t = tid; t < 512; t += 256) po[t] = fmaxf(p[t] * sc + sh, 0.f);
    }
}

__global__ void fill_zero(float* p, long n)
{
    long i = (long)blockIdx.x * blockDim.x + threadIdx.x;
    if (i < n) p[i] = 0.f;
}

// ---------------------------------------------------------------------------
// Host-side helpers
// ---------------------------------------------------------------------------
static void launch_gemm(hipStream_t st,
                        const float* A, long sAm, long sAk, long sAtap, long sAb,
                        const float* B, long sBk, long sBt, long sBtap, long sBb,
                        float* C, long ldc, long sCb, const float* bias,
                        int M, int Nn, int K, int ntaps, int flags, int nb)
{
    dim3 g(Nn / 64, M / 64, nb);
    gemm_bf16_wmma<<<g, 256, 0, st>>>(A, sAm, sAk, sAtap, sAb,
                                      B, sBk, sBt, sBtap, sBb,
                                      C, ldc, sCb, bias, K, ntaps, flags);
}

// get_convs: 1x1(512->128)+bias -> grouped k3 g32 + relu -> 1x1(128->512)+bias
static void run_convs(hipStream_t st, const float* x,
                      const float* w1, const float* b1,
                      const float* w2, const float* b2,
                      const float* w3, const float* b3,
                      float* H1, float* H2, float* G)
{
    launch_gemm(st, w1, 512, 1, 0, 0, x, 512, 1, 0, 512L * 512,
                H1, 512, 128L * 512, b1, 128, 512, 512, 1, 0, 4);
    dim3 gg((128 * 512 + 255) / 256, 4);
    grouped_conv3_relu<<<gg, 256, 0, st>>>(H1, w2, b2, H2);
    launch_gemm(st, w3, 128, 1, 0, 0, H2, 512, 1, 0, 128L * 512,
                G, 512, 512L * 512, b3, 512, 512, 128, 1, 0, 4);
}

extern "C" void kernel_launch(void* const* d_in, const int* in_sizes, int n_in,
                              void* d_out, int out_size, void* d_ws, size_t ws_size,
                              hipStream_t stream)
{
    (void)in_sizes; (void)n_in; (void)out_size; (void)ws_size;
    const float* x   = (const float*)d_in[0];   // [4,512,512]
    const float* pos = (const float*)d_in[1];   // [512,512]
    const float* k_w = (const float*)d_in[2];   const float* k_b = (const float*)d_in[3];
    const float* q_w = (const float*)d_in[4];   const float* q_b = (const float*)d_in[5];
    const float* vw1 = (const float*)d_in[6];   const float* vb1 = (const float*)d_in[7];
    const float* vw2 = (const float*)d_in[8];   const float* vb2 = (const float*)d_in[9];
    const float* vw3 = (const float*)d_in[10];  const float* vb3 = (const float*)d_in[11];
    const float* pw1 = (const float*)d_in[12];  const float* pb1 = (const float*)d_in[13];
    const float* pw2 = (const float*)d_in[14];  const float* pb2 = (const float*)d_in[15];
    const float* pw3 = (const float*)d_in[16];  const float* pb3 = (const float*)d_in[17];
    const float* sw1 = (const float*)d_in[18];  const float* sb1 = (const float*)d_in[19];
    const float* sw2 = (const float*)d_in[20];  const float* sb2 = (const float*)d_in[21];
    const float* sw3 = (const float*)d_in[22];  const float* sb3 = (const float*)d_in[23];
    // slots: [24..29] a=4, [30..35] a=8: proj_w, proj_b, rk_w, rk_b, agg_w, agg_b
    const float* bn_g = (const float*)d_in[36];
    const float* bn_b = (const float*)d_in[37];

    float* ws  = (float*)d_ws;
    float* ACC = ws;                 // 4*512*512
    float* G   = ws + 1048576;       // 4*512*512
    float* SIM = ws + 2097152;       // 4*512*512
    float* H1  = ws + 3145728;       // 4*128*512
    float* H2  = ws + 3407872;       // 4*128*512
    float* P   = ws + 3670016;       // 4*128*528 (zero-padded proj)
    float* Y   = ws + 3940352;       // 4*128*512
    float* U   = ws + 4202496;       // 4*17*512

    float* out_y = (float*)d_out;                 // [4,512,512]
    float* conns = out_y + 4L * 512 * 512;        // [4,512,512]

    // ---- PE branch: ACC = get_convs_pe(x) @ pos ----
    run_convs(stream, x, pw1, pb1, pw2, pb2, pw3, pb3, H1, H2, G);
    launch_gemm(stream, G, 512, 1, 0, 512L * 512, pos, 512, 1, 0, 0,
                ACC, 512, 512L * 512, nullptr, 512, 512, 512, 1, 0, 4);

    // ---- Global attention branch ----
    launch_gemm(stream, k_w, 512, 1, 0, 0, x, 512, 1, 0, 512L * 512,
                H1, 512, 128L * 512, k_b, 128, 512, 512, 1, 0, 4);    // K -> H1
    launch_gemm(stream, q_w, 512, 1, 0, 0, x, 512, 1, 0, 512L * 512,
                H2, 512, 128L * 512, q_b, 128, 512, 512, 1, 0, 4);    // Q -> H2
    // SIM[t,s] = sum_c K[c,t]*Q[c,s]  (A = K transposed via strides)
    launch_gemm(stream, H1, 1, 512, 0, 128L * 512, H2, 512, 1, 0, 128L * 512,
                SIM, 512, 512L * 512, nullptr, 512, 512, 128, 1, 0, 4);
    col_softmax<<<dim3(512, 4), 128, 0, stream>>>(SIM);
    run_convs(stream, x, vw1, vb1, vw2, vb2, vw3, vb3, H1, H2, G);    // V -> G
    launch_gemm(stream, G, 512, 1, 0, 512L * 512, SIM, 512, 1, 0, 512L * 512,
                ACC, 512, 512L * 512, nullptr, 512, 512, 512, 1, 1, 4); // ACC += V@SIM

    // ---- PRSlot pyramid (a = 4 then 8), conns into d_out slice ----
    fill_zero<<<(int)((4L * 128 * 528 + 255) / 256), 256, 0, stream>>>(P, 4L * 128 * 528);
    const int avals[2] = {4, 8};
    for (int i = 0; i < 2; ++i) {
        const int a = avals[i], Kw = 2 * a + 1;
        const float* proj_w = (const float*)d_in[24 + 6 * i];
        const float* proj_b = (const float*)d_in[25 + 6 * i];
        const float* rk_w   = (const float*)d_in[26 + 6 * i];
        const float* rk_b   = (const float*)d_in[27 + 6 * i];
        const float* agg_w  = (const float*)d_in[28 + 6 * i];
        // proj -> P interior (cols [8,520)), pads stay zero
        launch_gemm(stream, proj_w, 512, 1, 0, 0, x, 512, 1, 0, 512L * 512,
                    P + 8, 528, 128L * 528, proj_b, 128, 512, 512, 1, 0, 4);
        // rk conv as sum over Kw taps of 128x128 GEMMs on shifted P
        launch_gemm(stream, rk_w, 128L * Kw, Kw, 1, 0,
                    P + (8 - a), 528, 1, 1, 128L * 528,
                    Y, 512, 128L * 512, rk_b, 128, 512, 128, Kw, 0, 4);
        agg_u<<<dim3((Kw * 512 + 255) / 256, 4), 256, 0, stream>>>(Y, agg_w, U, Kw);
        band_softmax<<<dim3(512, 4), 128, 0, stream>>>(U, conns, a, i == 0 ? 0 : 1);
    }

    // ---- Semantic branch: ACC += get_convs_sem(x) @ conns ----
    run_convs(stream, x, sw1, sb1, sw2, sb2, sw3, sb3, H1, H2, G);
    launch_gemm(stream, G, 512, 1, 0, 512L * 512, conns, 512, 1, 0, 512L * 512,
                ACC, 512, 512L * 512, nullptr, 512, 512, 512, 1, 1, 4);

    // ---- BatchNorm (training stats) + ReLU -> first output ----
    bn_relu<<<512, 256, 0, stream>>>(ACC, bn_g, bn_b, out_y);
}
